// GRU1_2774548873456
// MI455X (gfx1250) — matrix operand order repaired and checked by reading, not compiled, not run on hardware
//
#include <hip/hip_runtime.h>
#include <hip/hip_bf16.h>
#include <math.h>

// ---------------------------------------------------------------------------
// GRU on MI455X (gfx1250, wave32, WMMA, TDM).
// Phase 1: x-projections as a large bf16 WMMA GEMM (whole GPU, L2-resident X).
// Phase 2: persistent 16-WG scan; per step the Tensor Data Mover copies the
//          h_prev matrix (32 KB) into LDS once per WG (s_wait_tensorcnt),
//          W_h slices are LDS-resident, gates computed with WMMA (f32 acc),
//          f32 element-wise update, device-scope arrive barrier per step.
// ---------------------------------------------------------------------------

typedef __attribute__((ext_vector_type(16))) __bf16 v16bf;
typedef __attribute__((ext_vector_type(8)))  __bf16 v8bf;
typedef __attribute__((ext_vector_type(8)))  float  v8f;
typedef unsigned int v4u __attribute__((ext_vector_type(4)));
typedef int          v8i __attribute__((ext_vector_type(8)));
typedef int          v4i __attribute__((ext_vector_type(4)));

#define T_STEPS 2048
#define BATCH   64
#define IDIM    256
#define HDIM    256
#define MROWS   (T_STEPS * BATCH)   // 131072

#if defined(__has_builtin)
#if __has_builtin(__builtin_amdgcn_tensor_load_to_lds) && __has_builtin(__builtin_amdgcn_s_wait_tensorcnt)
#define HAVE_TDM 1
#endif
#endif

// ---- WMMA fragment loaders (bf16, 16x16x32) --------------------------------
// A 16x32 (MxK): lane<16 -> row=lane, K 0..15 contiguous; lane>=16 -> K 16..31
__device__ __forceinline__ v16bf load_a_frag_g(const __bf16* __restrict__ base,
                                               int ldk, int mBase, int kBase, int lane) {
    const int half = lane >> 4;
    const int row  = mBase + (lane & 15);
    const __bf16* p = base + (size_t)row * ldk + kBase + half * 16;
    v8bf lo = *(const v8bf*)(p);
    v8bf hi = *(const v8bf*)(p + 8);
    v16bf a;
#pragma unroll
    for (int i = 0; i < 8; ++i) { a[i] = lo[i]; a[i + 8] = hi[i]; }
    return a;
}

// B 32x16 (KxN): lane<16 -> col=lane, K 0..15 contiguous; lane>=16 -> K 16..31
__device__ __forceinline__ v16bf load_b_frag_g(const __bf16* __restrict__ base,
                                               int ldk, int nBase, int kBase, int lane) {
    const int half = lane >> 4;
    const int col  = nBase + (lane & 15);
    return *(const v16bf*)(base + (size_t)col * ldk + kBase + half * 16);
}

// ---- f32 -> bf16 conversion -------------------------------------------------
__global__ void cvt_f32_to_bf16(const float* __restrict__ src, __bf16* __restrict__ dst, int n) {
    int i = blockIdx.x * 256 + threadIdx.x;
    if (i < n) dst[i] = (__bf16)src[i];
}

// ---- Phase 1: Xg[g] = X @ W_i[g]^T  (bf16 in, f32 acc, bf16 out) -----------
// One wave computes a 16(M) x 64(N) tile: 4 accumulators share each A fragment.
__global__ void __launch_bounds__(256)
xproj_kernel(const __bf16* __restrict__ Xb,    // [MROWS][IDIM]
             const __bf16* __restrict__ Wib,   // [3][HDIM][IDIM]
             __bf16* __restrict__ Xg)          // [3][MROWS][HDIM]
{
    const int lane = threadIdx.x & 31;
    const unsigned wid = blockIdx.x * 8u + (threadIdx.x >> 5);
    const int g  = (int)(wid % 3u);
    const unsigned rest = wid / 3u;
    const int nb = (int)(rest & 3u);    // 64-wide N block
    const int mt = (int)(rest >> 2);    // 16-wide M tile, 0..8191
    const int mBase = mt * 16;
    const int nBase = nb * 64;

    const __bf16* W = Wib + (size_t)g * HDIM * IDIM;
    v8f acc[4] = { {}, {}, {}, {} };

#pragma unroll
    for (int kk = 0; kk < 8; ++kk) {
        const int kBase = kk * 32;
        v16bf a = load_a_frag_g(Xb, IDIM, mBase, kBase, lane);
#pragma unroll
        for (int j = 0; j < 4; ++j) {
            v16bf b = load_b_frag_g(W, IDIM, nBase + j * 16, kBase, lane);
            acc[j] = __builtin_amdgcn_wmma_f32_16x16x32_bf16(
                false, a, false, b, (short)0, acc[j], false, false);
        }
    }

    __bf16* out = Xg + (size_t)g * MROWS * HDIM;
    const int half = lane >> 4, col = lane & 15;
#pragma unroll
    for (int j = 0; j < 4; ++j)
#pragma unroll
        for (int i = 0; i < 8; ++i) {
            const int row = mBase + i + half * 8;
            out[(size_t)row * HDIM + nBase + j * 16 + col] = (__bf16)acc[j][i];
        }
}

// ---- device-scope barrier for the 16 persistent workgroups -----------------
__device__ __forceinline__ void grid_barrier(unsigned* bar, unsigned target) {
    __threadfence();                 // release: all lanes' stores device-visible
    __syncthreads();
    if (threadIdx.x == 0) {
        __hip_atomic_fetch_add(bar, 1u, __ATOMIC_ACQ_REL, __HIP_MEMORY_SCOPE_AGENT);
        while (__hip_atomic_load(bar, __ATOMIC_ACQUIRE, __HIP_MEMORY_SCOPE_AGENT) < target)
            __builtin_amdgcn_s_sleep(2);
    }
    __syncthreads();
}

__device__ __forceinline__ float sigmoidf_fast(float x) {
    return 1.0f / (1.0f + __expf(-x));
}

// Dynamic-LDS partition for the scan kernel (72 KB total, WGP has 320 KB).
#define HMAT_OFF 0                         // bf16 [64][256]   h_prev tile  32 KB
#define WH_OFF   32768                     // bf16 [3][16][256] W_h slices  24 KB
#define ACC_OFF  57344                     // f32  [3][64][16]  gate sums   12 KB
#define HST_OFF  69632                     // f32  [64][16]     h state      4 KB
#define SMEM_BYTES 73728

// ---- Phase 2: persistent recurrent scan ------------------------------------
// 16 WGs x 256 threads (8 waves). WG nb owns h columns [16*nb, 16*nb+16).
// Per step: TDM copies h_prev (64x256 bf16) into LDS once; 12 tile-tasks
// (3 gates x 4 M-tiles) of 8 chained WMMAs each, A+B from LDS; f32 update.
__global__ void __launch_bounds__(256)
gru_scan_kernel(const __bf16* __restrict__ Whb,   // [3][HDIM][HDIM]
                const __bf16* __restrict__ Xg,    // [3][MROWS][HDIM]
                const float* __restrict__ b_ir, const float* __restrict__ b_iz,
                const float* __restrict__ b_in, const float* __restrict__ b_hr,
                const float* __restrict__ b_hz, const float* __restrict__ b_hn,
                __bf16* __restrict__ hbuf0, __bf16* __restrict__ hbuf1,
                float* __restrict__ out, unsigned* __restrict__ bar)
{
    extern __shared__ __attribute__((aligned(128))) char smem[];
    __bf16* lds_hmat = (__bf16*)(smem + HMAT_OFF);   // [64][256]
    __bf16* lds_wh   = (__bf16*)(smem + WH_OFF);     // [(g*16+n)*256 + k]
    float*  lds_acc  = (float*) (smem + ACC_OFF);    // [(g*64+row)*16 + col]
    float*  lds_hst  = (float*) (smem + HST_OFF);    // [row*16 + col]

    const int tid   = threadIdx.x;
    const int lane  = tid & 31;
    const int wv    = tid >> 5;
    const int nb    = blockIdx.x;       // 0..15
    const int nBase = nb * 16;

    // Stage this WG's 3x[16 x 256] bf16 weight slices into LDS (once).
    for (int idx = tid; idx < 3 * 16 * HDIM; idx += 256) {
        const int g = idx / (16 * HDIM);
        const int r = idx % (16 * HDIM);
        const int n = r / HDIM;
        const int k = r % HDIM;
        lds_wh[(g * 16 + n) * HDIM + k] = Whb[((size_t)g * HDIM + nBase + n) * HDIM + k];
    }
    // Zero h state (own slice of global bf16 h0 + private f32 slice).
#pragma unroll
    for (int u = 0; u < 4; ++u) {
        const int e = tid + u * 256;
        const int row = e >> 4, col = e & 15;
        hbuf0[row * HDIM + nBase + col] = (__bf16)0.0f;
        lds_hst[row * 16 + col] = 0.0f;
    }
    unsigned barcnt = 16u;
    grid_barrier(bar, barcnt);          // all h0 slices visible

    const unsigned lds_hmat_addr = (unsigned)(size_t)lds_hmat; // flat low 32b = LDS byte offset

    for (int t = 0; t < T_STEPS; ++t) {
        const __bf16* cur = (t & 1) ? hbuf1 : hbuf0;
        __bf16*       nxt = (t & 1) ? hbuf0 : hbuf1;

        // --- Pull h_prev (32 KB, contiguous) into LDS via the Tensor Data
        //     Mover: one DMA per WG instead of 3x redundant per-wave loads.
        if (wv == 0) {
#ifdef HAVE_TDM
            const unsigned long long ga = (unsigned long long)(size_t)cur;
            v4u g0;                                  // D# group 0 (ISA 8.3)
            g0[0] = 1u;                              // count=1, user mode
            g0[1] = lds_hmat_addr;                   // lds_addr (bytes)
            g0[2] = (unsigned)(ga & 0xFFFFFFFFull);  // global_addr[31:0]
            g0[3] = (unsigned)((ga >> 32) & 0x01FFFFFFull) | 0x80000000u; // addr[56:32] | type=2
            v8i g1;                                  // D# group 1 (ISA 8.4)
            g1[0] = (int)(3u << 16);                 // mask=0, data_size=3 (8B units)
            g1[1] = (int)(4096u << 16);              // tensor_dim0 = 4096 units (32 KB)
            g1[2] = (int)(1u << 16);                 // tensor_dim0 hi=0, tensor_dim1=1
            g1[3] = (int)(4096u << 16);              // tile_dim0 = 4096 units
            g1[4] = 0;                               // tile_dim1/2 unused
            g1[5] = (int)4096u;                      // tensor_dim0_stride = 4096 units
            g1[6] = 0;
            g1[7] = 0;
            v4i gz4 = { 0, 0, 0, 0 };                // groups 2/3 unused (<=2D)
            v8i gz8 = { 0, 0, 0, 0, 0, 0, 0, 0 };    // trailing group (this toolchain's 6-arg form)
            __builtin_amdgcn_tensor_load_to_lds(g0, g1, gz4, gz4, gz8, 0);
            __builtin_amdgcn_s_wait_tensorcnt((short)0);
#else
            // Fallback: cooperative wide copy (one wave, b128 chunks).
            const v8bf* src = (const v8bf*)cur;
            v8bf* dst = (v8bf*)lds_hmat;
            for (int i = lane; i < (BATCH * HDIM) / 8; i += 32) dst[i] = src[i];
#endif
        }
        __syncthreads();                 // LDS h_prev visible to all waves

        // 12 tasks over 8 waves: wave w takes tasks {w, w+8 (if <12)}.
        for (int task = wv; task < 12; task += 8) {
            const int g  = task >> 2;          // gate 0..2 (r,z,n)
            const int mt = task & 3;           // 16-row M tile of the batch
            v8f acc = {};
#pragma unroll
            for (int kk = 0; kk < 8; ++kk) {
                const int kBase = kk * 32;
                // A: 16x32 tile of h_prev from LDS
                const int half = lane >> 4;
                const int row  = mt * 16 + (lane & 15);
                const __bf16* pa = lds_hmat + row * HDIM + kBase + half * 16;
                v8bf lo = *(const v8bf*)(pa);
                v8bf hi = *(const v8bf*)(pa + 8);
                v16bf a;
#pragma unroll
                for (int i = 0; i < 8; ++i) { a[i] = lo[i]; a[i + 8] = hi[i]; }
                // B: 32x16 slice of W_h[g] from LDS
                v16bf b = *(const v16bf*)(lds_wh + ((g * 16 + (lane & 15)) * HDIM) + kBase + half * 16);
                acc = __builtin_amdgcn_wmma_f32_16x16x32_bf16(
                    false, a, false, b, (short)0, acc, false, false);
            }
#pragma unroll
            for (int i = 0; i < 8; ++i)
                lds_acc[(g * 64 + mt * 16 + i + (lane >> 4) * 8) * 16 + (lane & 15)] = acc[i];
        }
        __syncthreads();

        // Element-wise GRU update in f32: 1024 elems, 4 per thread.
#pragma unroll
        for (int u = 0; u < 4; ++u) {
            const int e = tid + u * 256;
            const int row = e >> 4, col = e & 15;
            const int hcol = nBase + col;
            const size_t xoff = ((size_t)t * BATCH + row) * HDIM + hcol;
            const float xr = (float)Xg[0 * (size_t)MROWS * HDIM + xoff];
            const float xz = (float)Xg[1 * (size_t)MROWS * HDIM + xoff];
            const float xn = (float)Xg[2 * (size_t)MROWS * HDIM + xoff];
            const float rs = lds_acc[(0 * 64 + row) * 16 + col];
            const float zs = lds_acc[(1 * 64 + row) * 16 + col];
            const float ns = lds_acc[(2 * 64 + row) * 16 + col];
            const float r = sigmoidf_fast(xr + b_ir[hcol] + rs + b_hr[hcol]);
            const float z = sigmoidf_fast(xz + b_iz[hcol] + zs + b_hz[hcol]);
            const float n = tanhf(xn + b_in[hcol] + r * (ns + b_hn[hcol]));
            const float hp = lds_hst[row * 16 + col];
            const float hv = (1.0f - z) * n + z * hp;
            lds_hst[row * 16 + col] = hv;
            out[xoff] = hv;                       // f32 result
            nxt[row * HDIM + hcol] = (__bf16)hv;  // bf16 for next step's GEMMs
            // Prefetch next step's x-projections (L2-resident, predictable).
            if (t + 1 < T_STEPS) {
                const size_t nxoff = xoff + (size_t)BATCH * HDIM;
                __builtin_prefetch((const void*)(Xg + 0 * (size_t)MROWS * HDIM + nxoff), 0, 1);
                __builtin_prefetch((const void*)(Xg + 1 * (size_t)MROWS * HDIM + nxoff), 0, 1);
                __builtin_prefetch((const void*)(Xg + 2 * (size_t)MROWS * HDIM + nxoff), 0, 1);
            }
        }
        barcnt += 16u;
        grid_barrier(bar, barcnt);
    }
}

// ---------------------------------------------------------------------------
extern "C" void kernel_launch(void* const* d_in, const int* in_sizes, int n_in,
                              void* d_out, int out_size, void* d_ws, size_t ws_size,
                              hipStream_t stream) {
    (void)in_sizes; (void)n_in; (void)out_size; (void)ws_size;

    const float* X = (const float*)d_in[0];
    const float* W_i[3] = { (const float*)d_in[1], (const float*)d_in[2], (const float*)d_in[3] };
    const float* W_h[3] = { (const float*)d_in[4], (const float*)d_in[5], (const float*)d_in[6] };
    const float* b_ir = (const float*)d_in[7];
    const float* b_iz = (const float*)d_in[8];
    const float* b_in = (const float*)d_in[9];
    const float* b_hr = (const float*)d_in[10];
    const float* b_hz = (const float*)d_in[11];
    const float* b_hn = (const float*)d_in[12];

    // Workspace layout (~270 MB total)
    char* ws = (char*)d_ws;
    size_t off = 0;
    __bf16* Xb  = (__bf16*)(ws + off); off += (size_t)MROWS * IDIM * 2;      //  67 MB
    __bf16* Wib = (__bf16*)(ws + off); off += (size_t)3 * HDIM * IDIM * 2;   // 384 KB
    __bf16* Whb = (__bf16*)(ws + off); off += (size_t)3 * HDIM * HDIM * 2;   // 384 KB
    __bf16* Xg  = (__bf16*)(ws + off); off += (size_t)3 * MROWS * HDIM * 2;  // 201 MB
    __bf16* h0  = (__bf16*)(ws + off); off += (size_t)BATCH * HDIM * 2;
    __bf16* h1  = (__bf16*)(ws + off); off += (size_t)BATCH * HDIM * 2;
    unsigned* bar = (unsigned*)(ws + off); off += 64;

    (void)hipMemsetAsync(bar, 0, 64, stream);

    // Convert inputs/weights to bf16.
    {
        const int n = MROWS * IDIM;
        cvt_f32_to_bf16<<<(n + 255) / 256, 256, 0, stream>>>(X, Xb, n);
    }
    for (int g = 0; g < 3; ++g) {
        const int n = HDIM * IDIM;
        cvt_f32_to_bf16<<<(n + 255) / 256, 256, 0, stream>>>(W_i[g], Wib + (size_t)g * n, n);
        cvt_f32_to_bf16<<<(n + 255) / 256, 256, 0, stream>>>(W_h[g], Whb + (size_t)g * n, n);
    }

    // Phase 1: 98304 waves -> 12288 blocks of 8 waves.
    xproj_kernel<<<12288, 256, 0, stream>>>(Xb, Wib, Xg);

    // Phase 2: persistent scan, 16 WGs, 72 KB dynamic LDS each.
    gru_scan_kernel<<<16, 256, SMEM_BYTES, stream>>>(Whb, Xg,
                                                     b_ir, b_iz, b_in, b_hr, b_hz, b_hn,
                                                     h0, h1, (float*)d_out, bar);
}
